// CombSub_5403068858478
// MI455X (gfx1250) — matrix operand order repaired and checked by reading, not compiled, not run on hardware
//
#include <hip/hip_runtime.h>
#include <cstdint>
#include <cstddef>

// ---------------------------------------------------------------- constants
#define PI_F 3.14159265358979f

constexpr float SR_F   = 44100.0f;
constexpr int   BLK    = 512;
constexpr int   NF     = 2400;
constexpr int   NB     = 8;
constexpr int   T_LEN  = NF * BLK;      // 1,228,800
constexpr int   MROWS  = NB * NF;       // 19,200
constexpr int   HID    = 256;
constexpr int   NOUT   = 1024;          // 256 allpass + 512 harm + 256 noise

// WMMA vector types (gfx1250, wave32)
typedef _Float16 v16h __attribute__((ext_vector_type(16)));
typedef _Float16 v8h  __attribute__((ext_vector_type(8)));
typedef _Float16 v4h  __attribute__((ext_vector_type(4)));
typedef float    v8f  __attribute__((ext_vector_type(8)));

// ---------------------------------------------------------------- utilities
__global__ void zero_kernel(float* __restrict__ p, size_t n) {
    size_t i = (size_t)blockIdx.x * blockDim.x + threadIdx.x;
    if (i < n) p[i] = 0.0f;
}

// cos/sin tables for inverse real DFT of the filter spectra.
__global__ void table510_kernel(float* __restrict__ ct, float* __restrict__ st) {
    int i = blockIdx.x * blockDim.x + threadIdx.x;
    if (i >= 256 * 510) return;
    int k = i / 510, t = i % 510;
    int m = (k * t) % 510;
    float ang = 2.0f * PI_F * (float)m / 510.0f;
    ct[i] = cosf(ang);
    st[i] = sinf(ang);
}
__global__ void table1022_kernel(float* __restrict__ ct) {
    int i = blockIdx.x * blockDim.x + threadIdx.x;
    if (i >= 512 * 1022) return;
    int k = i / 1022, t = i % 1022;
    int m = (k * t) % 1022;
    ct[i] = cosf(2.0f * PI_F * (float)m / 1022.0f);
}

// --------------------------------------------------- f0 phase prefix (scan)
__global__ void scan_kernel(const float* __restrict__ f0f, float* __restrict__ S) {
    const int b = blockIdx.x;
    __shared__ float fs[256];
    __shared__ float runS;
    if (threadIdx.x == 0) runS = 0.0f;
    __syncthreads();
    for (int c0 = 0; c0 < NF; c0 += 256) {
        int n = c0 + threadIdx.x;
        float v = 0.0f;
        if (n < NF) {
            float a  = f0f[b * NF + n];
            float a1 = f0f[b * NF + (n + 1 < NF ? n + 1 : NF - 1)];
            v = (512.0f * a + 255.5f * (a1 - a)) * (1.0f / SR_F);
        }
        fs[threadIdx.x] = v;
        __syncthreads();
        if (threadIdx.x == 0) {
            float r = runS;
            for (int i = 0; i < 256 && (c0 + i) < NF; ++i) {
                S[b * (NF + 1) + c0 + i] = r;
                r += fs[i];
            }
            runS = r;
        }
        __syncthreads();
    }
    if (threadIdx.x == 0) S[b * (NF + 1) + NF] = runS;
}

__global__ void phase_kernel(const float* __restrict__ f0f, const float* __restrict__ S,
                             float* __restrict__ phase_out) {
    int i = blockIdx.x * blockDim.x + threadIdx.x;
    if (i >= MROWS) return;
    int b = i / NF, n = i % NF;
    float x = S[b * (NF + 1) + n] + f0f[i] * (1.0f / SR_F);
    x -= rintf(x);
    phase_out[i] = 2.0f * PI_F * x;
}

__global__ void combtooth_kernel(const float* __restrict__ f0f, const float* __restrict__ S,
                                 float* __restrict__ comb) {
    size_t gid = (size_t)blockIdx.x * blockDim.x + threadIdx.x;
    if (gid >= (size_t)NB * T_LEN) return;
    int b = (int)(gid / T_LEN);
    int s = (int)(gid % T_LEN);
    int n = s >> 9, k = s & 511;
    float f0n  = f0f[b * NF + n];
    float f0n1 = f0f[b * NF + (n + 1 < NF ? n + 1 : NF - 1)];
    float d = f0n1 - f0n;
    float kk = (float)k;
    float f0u = f0n + d * kk * (1.0f / 512.0f);
    float x = S[b * (NF + 1) + n] +
              ((kk + 1.0f) * f0n + d * (kk * (kk + 1.0f) * 0.5f) * (1.0f / 512.0f)) * (1.0f / SR_F);
    x -= rintf(x);
    float v = SR_F * x / (f0u + 1e-3f);
    float ct = (fabsf(v) < 1e-6f) ? 1.0f : sinf(PI_F * v) / (PI_F * v);
    comb[gid] = ct;
}

// ----------------------------------------------------------- WMMA MLP GEMMs
// One wave = one 16x16 f32 tile, full K=256 staged in LDS.
// A staged row-major As[16][256] f16; B staged TRANSPOSED Bt[16(n)][256(k)] f16 so
// every lane's fragment is contiguous 16B chunks -> ds_load_b128.
//
// A frag (16x32 f16, K-step j): lanes 0-15 M=lane, halves [j*32 .. +7] and [j*32+16 .. +23];
// lanes 16-31 M=lane-16, halves [j*32+8 ..] and [j*32+24 ..].
// B frag (32x16): N = lane&15; same K order as A: lanes 0-15 K=j*32+0..7,16..23, lanes 16-31 K=j*32+8..15,24..31.
// C/D (16x16 f32): VGPR r -> M=r (lanes 0-15) / 8+r (lanes 16-31), N = lane&15.
__device__ inline v16h frag_a_lds(const _Float16* As, int lane, int j) {
    int m    = (lane < 16) ? lane : lane - 16;
    int base = j * 32 + ((lane < 16) ? 0 : 8);
    v8h lo = *(const v8h*)(As + m * 256 + base);
    v8h hi = *(const v8h*)(As + m * 256 + base + 16);
    return __builtin_shufflevector(lo, hi, 0, 1, 2, 3, 4, 5, 6, 7,
                                   8, 9, 10, 11, 12, 13, 14, 15);
}
__device__ inline v16h frag_b_lds(const _Float16* Bt, int lane, int j) {
    int n    = lane & 15;
    int kofs = j * 32 + ((lane < 16) ? 0 : 8);
    v8h lo = *(const v8h*)(Bt + n * 256 + kofs);
    v8h hi = *(const v8h*)(Bt + n * 256 + kofs + 16);
    return __builtin_shufflevector(lo, hi, 0, 1, 2, 3, 4, 5, 6, 7,
                                   8, 9, 10, 11, 12, 13, 14, 15);
}

// GEMM1: h = units @ W_unit + rank1(f0,phase,vol) + b_unit ; tanh -> f16
__global__ void mlp1_kernel(const float* __restrict__ units,   // [M x 256]
                            const float* __restrict__ Wu,      // [256 x 256]
                            const float* __restrict__ bu,
                            const float* __restrict__ Wf0,
                            const float* __restrict__ Wph,
                            const float* __restrict__ Wv,
                            const float* __restrict__ f0f,
                            const float* __restrict__ volf,
                            const float* __restrict__ phase,
                            _Float16* __restrict__ th)         // [M x 256]
{
    const int lane = threadIdx.x;
    const int gm0 = blockIdx.x * 16;
    const int gn0 = blockIdx.y * 16;
    __shared__ _Float16 As[16 * 256];
    __shared__ _Float16 Bt[16 * 256];

    // A: 4096 floats, coalesced float4 loads -> packed v4h LDS stores
#pragma unroll 4
    for (int it = 0; it < 32; ++it) {
        int fid  = it * 32 + lane;        // 0..1023 float4s
        int base = fid * 4;
        int r = base >> 8, c = base & 255;
        const float4 q = *(const float4*)(units + (size_t)(gm0 + r) * 256 + c);
        v4h h4 = {(_Float16)q.x, (_Float16)q.y, (_Float16)q.z, (_Float16)q.w};
        *(v4h*)(As + r * 256 + c) = h4;
    }
    // B: coalesced along N, scattered (transposed) into Bt[n][k]
#pragma unroll 4
    for (int it = 0; it < 32; ++it) {
        int fid = it * 32 + lane;         // k = fid>>2 (0..255), n4 = (fid&3)*4
        int k = fid >> 2, n4 = (fid & 3) * 4;
        const float4 q = *(const float4*)(Wu + (size_t)k * 256 + gn0 + n4);
        Bt[(n4 + 0) * 256 + k] = (_Float16)q.x;
        Bt[(n4 + 1) * 256 + k] = (_Float16)q.y;
        Bt[(n4 + 2) * 256 + k] = (_Float16)q.z;
        Bt[(n4 + 3) * 256 + k] = (_Float16)q.w;
    }
    __syncthreads();

    v8f acc = {};
#pragma unroll
    for (int j = 0; j < 8; ++j) {
        v16h a = frag_a_lds(As, lane, j);
        v16h b = frag_b_lds(Bt, lane, j);
        acc = __builtin_amdgcn_wmma_f32_16x16x32_f16(false, a, false, b, (short)0, acc,
                                                     false, false);
    }

    const int n = lane & 15;
    const int gn = gn0 + n;
    float wf = Wf0[gn], wp = Wph[gn], wv = Wv[gn], bb = bu[gn];
#pragma unroll
    for (int r = 0; r < 8; ++r) {
        int m = (lane < 16) ? r : r + 8;
        int gm = gm0 + m;
        float lf0 = log1pf(f0f[gm] * (1.0f / 700.0f));
        float ph  = phase[gm] * (1.0f / PI_F);
        float h = acc[r] + lf0 * wf + ph * wp + volf[gm] * wv + bb;
        th[(size_t)gm * 256 + gn] = (_Float16)tanhf(h);
    }
}

// GEMM2: e = tanh_h @ W_out + b_out
__global__ void mlp2_kernel(const _Float16* __restrict__ A,   // [M x 256] f16
                            const float* __restrict__ Wo,     // [256 x 1024]
                            const float* __restrict__ bo,
                            float* __restrict__ eOut)         // [M x 1024]
{
    const int lane = threadIdx.x;
    const int gm0 = blockIdx.x * 16;
    const int gn0 = blockIdx.y * 16;
    __shared__ _Float16 As[16 * 256];
    __shared__ _Float16 Bt[16 * 256];

    // A already f16: 4096 halves = 512 v8h, 16 per lane
#pragma unroll 4
    for (int it = 0; it < 16; ++it) {
        int fid  = it * 32 + lane;        // 0..511 v8h chunks
        int base = fid * 8;
        int r = base >> 8, c = base & 255;
        *(v8h*)(As + r * 256 + c) = *(const v8h*)(A + (size_t)(gm0 + r) * 256 + c);
    }
#pragma unroll 4
    for (int it = 0; it < 32; ++it) {
        int fid = it * 32 + lane;
        int k = fid >> 2, n4 = (fid & 3) * 4;
        const float4 q = *(const float4*)(Wo + (size_t)k * 1024 + gn0 + n4);
        Bt[(n4 + 0) * 256 + k] = (_Float16)q.x;
        Bt[(n4 + 1) * 256 + k] = (_Float16)q.y;
        Bt[(n4 + 2) * 256 + k] = (_Float16)q.z;
        Bt[(n4 + 3) * 256 + k] = (_Float16)q.w;
    }
    __syncthreads();

    v8f acc = {};
#pragma unroll
    for (int j = 0; j < 8; ++j) {
        v16h a = frag_a_lds(As, lane, j);
        v16h b = frag_b_lds(Bt, lane, j);
        acc = __builtin_amdgcn_wmma_f32_16x16x32_f16(false, a, false, b, (short)0, acc,
                                                     false, false);
    }

    const int n = lane & 15;
    const int gn = gn0 + n;
    float bb = bo[gn];
#pragma unroll
    for (int r = 0; r < 8; ++r) {
        int m = (lane < 16) ? r : r + 8;
        eOut[(size_t)(gm0 + m) * 1024 + gn] = acc[r] + bb;
    }
}

// allpass spectrum: exp(i * cumsum(pi*tanh(gd)))
__global__ void allpass_kernel(const float* __restrict__ e,
                               float* __restrict__ apRe, float* __restrict__ apIm) {
    int row = blockIdx.x * blockDim.x + threadIdx.x;
    if (row >= MROWS) return;
    float acc = 0.0f;
    for (int c = 0; c < 256; ++c) {
        acc += PI_F * tanhf(e[(size_t)row * 1024 + c]);
        apRe[(size_t)row * 256 + c] = cosf(acc);
        apIm[(size_t)row * 256 + c] = sinf(acc);
    }
}

// -------------------------------------------------- FFT-based frame filters
__device__ inline float2 cmulc(float2 a, float2 b) {
    return make_float2(a.x * b.x - a.y * b.y, a.x * b.y + a.y * b.x);
}

template <int N>
__device__ float2* block_fft(float2* s, float2* d, float sign) {
    float2* src = s;
    float2* dst = d;
    for (int L = 1; L < N; L <<= 1) {
        for (int i = threadIdx.x; i < (N >> 1); i += blockDim.x) {
            int k = i & (L - 1);
            int j = ((i - k) << 1) + k;
            float ang = sign * PI_F * (float)k / (float)L;
            float sn, cs;
            __sincosf(ang, &sn, &cs);
            float2 u = src[i];
            float2 t = src[i + (N >> 1)];
            float2 v = make_float2(t.x * cs - t.y * sn, t.x * sn + t.y * cs);
            dst[j]     = make_float2(u.x + v.x, u.y + v.y);
            dst[j + L] = make_float2(u.x - v.x, u.y - v.y);
        }
        __syncthreads();
        float2* tmp = src; src = dst; dst = tmp;
    }
    return src;
}

// One block = one frame; async-LDS staged audio, per-frame IR via table DFT,
// packed-real FFT convolution, atomic overlap-add.
// FILT: 1 = allpass (complex spectrum, no window)
//       2 = harmonic shaping (exp(e), f0-dependent Hann)
//       3 = noise shaping (exp(e)/128, standard Hann, audio = 2*u-1)
template <int NFFT, int NIR, int NMAG, int FILT>
__global__ void conv_kernel(const float* __restrict__ audio,
                            const float* __restrict__ apRe,
                            const float* __restrict__ apIm,
                            const float* __restrict__ eBuf,
                            const float* __restrict__ ct,
                            const float* __restrict__ st,
                            const float* __restrict__ f0f,
                            float* __restrict__ out) {
    const int f = blockIdx.x, b = blockIdx.y;
    const int row = b * NF + f;
    const int tid = threadIdx.x;
    __shared__ float2 buf0[NFFT];
    __shared__ float2 buf1[NFFT];
    __shared__ float  irs[NIR];
    __shared__ float  stage[BLK];
    __shared__ float  Pk[(FILT == 1) ? 2 * NMAG : NMAG];

    // Kick off the audio frame -> LDS async copy (ASYNCcnt-tracked) so the DMA
    // overlaps the IR inverse-DFT below. 128 lanes x 16B = 2KB frame.
    const float* aud = audio + (size_t)b * T_LEN + (size_t)f * BLK;
    if (tid < BLK / 4) {
        uint32_t laddr = (uint32_t)(uintptr_t)(const void*)&stage[tid * 4];
        uint64_t gaddr = (uint64_t)(uintptr_t)(aud + tid * 4);
        asm volatile("global_load_async_to_lds_b128 %0, %1, off"
                     :: "v"(laddr), "v"(gaddr)
                     : "memory");
    }
    if (tid == 0) __builtin_prefetch(eBuf + (size_t)row * 1024, 0, 0);

    for (int k = tid; k < NMAG; k += blockDim.x) {
        if (FILT == 1) {
            Pk[k]        = apRe[(size_t)row * NMAG + k];
            Pk[NMAG + k] = apIm[(size_t)row * NMAG + k];
        } else if (FILT == 2) {
            Pk[k] = expf(eBuf[(size_t)row * 1024 + 256 + k]);
        } else {
            Pk[k] = expf(eBuf[(size_t)row * 1024 + 768 + k]) * (1.0f / 128.0f);
        }
    }
    __syncthreads();

    float hw = 0.0f;
    if (FILT == 2) hw = 1.5f * SR_F / (f0f[row] + 1e-3f);

    for (int i = tid; i < NIR; i += blockDim.x) {
        int torig = i + NIR / 2;
        if (torig >= NIR) torig -= NIR;   // fftshift folded into table index
        float acc = 0.0f;
        if (FILT == 1) {
            for (int k = 0; k < NMAG; ++k) {
                float g = (k == 0 || k == NMAG - 1) ? 1.0f : 2.0f;
                acc += g * (Pk[k] * ct[k * NIR + torig] - Pk[NMAG + k] * st[k * NIR + torig]);
            }
        } else {
            for (int k = 0; k < NMAG; ++k) {
                float g = (k == 0 || k == NMAG - 1) ? 1.0f : 2.0f;
                acc += g * Pk[k] * ct[k * NIR + torig];
            }
        }
        float ir = acc * (1.0f / (float)NIR);
        float tt = (float)i - (float)(NIR / 2);
        if (FILT == 2) {
            float r = tt / hw;
            ir *= (fabsf(r) < 1.0f) ? (0.5f + 0.5f * cosf(PI_F * r)) : 0.0f;
        } else if (FILT == 3) {
            ir *= 0.5f + 0.5f * cosf(PI_F * tt / (float)(NIR / 2));
        }
        irs[i] = ir;
    }
    // drain the async copy, then make stage/irs visible to all waves
    asm volatile("s_wait_asynccnt 0" ::: "memory");
    __syncthreads();

    for (int i = tid; i < NFFT; i += blockDim.x) {
        float re = 0.0f;
        if (i < BLK) re = (FILT == 3) ? (2.0f * stage[i] - 1.0f) : stage[i];
        float im = (i < NIR) ? irs[i] : 0.0f;
        buf0[i] = make_float2(re, im);
    }
    __syncthreads();

    float2* Z = block_fft<NFFT>(buf0, buf1, -1.0f);
    float2* O = (Z == buf0) ? buf1 : buf0;
    for (int k = tid; k <= NFFT / 2; k += blockDim.x) {
        int nk = (NFFT - k) & (NFFT - 1);
        float2 zk = Z[k], zn = Z[nk];
        float2 A  = make_float2(0.5f * (zk.x + zn.x), 0.5f * (zk.y - zn.y));
        float2 Bc = make_float2(0.5f * (zk.y + zn.y), 0.5f * (zn.x - zk.x));
        float2 C  = cmulc(A, Bc);
        O[k] = C;
        if (k > 0 && k < NFFT / 2) O[NFFT - k] = make_float2(C.x, -C.y);
    }
    __syncthreads();
    float2* R = block_fft<NFFT>(O, Z, 1.0f);

    const int seglen = BLK + NIR;
    const int delay  = NIR / 2;
    for (int i = tid; i < seglen; i += blockDim.x) {
        int s = f * BLK + i - delay;
        if (s >= 0 && s < T_LEN) {
            atomicAdd(&out[(size_t)b * T_LEN + s], R[i].x * (1.0f / (float)NFFT));
        }
    }
}

__global__ void final_add_kernel(const float* __restrict__ harm, const float* __restrict__ noi,
                                 float* __restrict__ sig) {
    size_t i = (size_t)blockIdx.x * blockDim.x + threadIdx.x;
    if (i < (size_t)NB * T_LEN) sig[i] = harm[i] + noi[i];
}

// ---------------------------------------------------------------- launcher
extern "C" void kernel_launch(void* const* d_in, const int* in_sizes, int n_in,
                              void* d_out, int out_size, void* d_ws, size_t ws_size,
                              hipStream_t stream) {
    (void)in_sizes; (void)n_in; (void)out_size; (void)ws_size;

    const float* units = (const float*)d_in[0];
    const float* f0f   = (const float*)d_in[1];
    const float* volf  = (const float*)d_in[2];
    const float* noise = (const float*)d_in[3];
    const float* Wu    = (const float*)d_in[4];
    const float* bu    = (const float*)d_in[5];
    const float* Wf0   = (const float*)d_in[6];
    const float* Wph   = (const float*)d_in[7];
    const float* Wv    = (const float*)d_in[8];
    const float* Wo    = (const float*)d_in[9];
    const float* bo    = (const float*)d_in[10];

    float* out_signal = (float*)d_out;
    float* out_phase  = out_signal + (size_t)NB * T_LEN;
    float* out_harm   = out_phase + (size_t)NB * NF;
    float* out_noise  = out_harm + (size_t)NB * T_LEN;

    size_t off = 0;
    auto carve = [&](size_t bytes) -> char* {
        char* p = (char*)d_ws + off;
        off = (off + bytes + 255) & ~(size_t)255;
        return p;
    };
    float*     S     = (float*)carve(sizeof(float) * NB * (NF + 1));
    float*     comb  = (float*)carve(sizeof(float) * (size_t)NB * T_LEN);
    _Float16*  th    = (_Float16*)carve(sizeof(_Float16) * (size_t)MROWS * 256);
    float*     eBuf  = (float*)carve(sizeof(float) * (size_t)MROWS * 1024);
    float*     apRe  = (float*)carve(sizeof(float) * (size_t)MROWS * 256);
    float*     apIm  = (float*)carve(sizeof(float) * (size_t)MROWS * 256);
    float*     htmp  = (float*)carve(sizeof(float) * (size_t)NB * T_LEN);
    float*     ct1   = (float*)carve(sizeof(float) * 256 * 510);
    float*     st1   = (float*)carve(sizeof(float) * 256 * 510);
    float*     ct2   = (float*)carve(sizeof(float) * 512 * 1022);

    const size_t BT = (size_t)NB * T_LEN;
    const int zgrid = (int)((BT + 255) / 256);

    zero_kernel<<<zgrid, 256, 0, stream>>>(htmp, BT);
    zero_kernel<<<zgrid, 256, 0, stream>>>(out_harm, BT);
    zero_kernel<<<zgrid, 256, 0, stream>>>(out_noise, BT);

    table510_kernel<<<(256 * 510 + 255) / 256, 256, 0, stream>>>(ct1, st1);
    table1022_kernel<<<(512 * 1022 + 255) / 256, 256, 0, stream>>>(ct2);

    scan_kernel<<<NB, 256, 0, stream>>>(f0f, S);
    phase_kernel<<<(MROWS + 255) / 256, 256, 0, stream>>>(f0f, S, out_phase);
    combtooth_kernel<<<zgrid, 256, 0, stream>>>(f0f, S, comb);

    mlp1_kernel<<<dim3(MROWS / 16, HID / 16), 32, 0, stream>>>(
        units, Wu, bu, Wf0, Wph, Wv, f0f, volf, out_phase, th);
    mlp2_kernel<<<dim3(MROWS / 16, NOUT / 16), 32, 0, stream>>>(th, Wo, bo, eBuf);

    allpass_kernel<<<(MROWS + 255) / 256, 256, 0, stream>>>(eBuf, apRe, apIm);

    conv_kernel<1024, 510, 256, 1><<<dim3(NF, NB), 256, 0, stream>>>(
        comb, apRe, apIm, eBuf, ct1, st1, f0f, htmp);
    conv_kernel<2048, 1022, 512, 2><<<dim3(NF, NB), 256, 0, stream>>>(
        htmp, apRe, apIm, eBuf, ct2, st1, f0f, out_harm);
    conv_kernel<1024, 510, 256, 3><<<dim3(NF, NB), 256, 0, stream>>>(
        noise, apRe, apIm, eBuf, ct1, st1, f0f, out_noise);

    final_add_kernel<<<zgrid, 256, 0, stream>>>(out_harm, out_noise, out_signal);
}
